// GNNEncoder_24515673325797
// MI455X (gfx1250) — matrix operand order repaired and checked by reading, not compiled, not run on hardware
//
#include <hip/hip_runtime.h>

typedef __attribute__((ext_vector_type(16))) __bf16 v16bf;
typedef __attribute__((ext_vector_type(8)))  float  v8f;

#define EPS_BN   1e-5f
#define EPS_NORM 1e-12f

__device__ __forceinline__ unsigned short f2bf(float f) {
  union { float f; unsigned u; } c; c.f = f;
  unsigned u = c.u;
  unsigned r = u + 0x7FFFu + ((u >> 16) & 1u);   // round-to-nearest-even
  return (unsigned short)(r >> 16);
}

union Frag16 {
  v16bf v;
  unsigned short s[16];
  uint4 q[2];
};

// ---------------------------------------------------------------- zero
__global__ void zero_kernel(float4* __restrict__ p, long n4) {
  long i = (long)blockIdx.x * blockDim.x + threadIdx.x;
  if (i < n4) p[i] = make_float4(0.f, 0.f, 0.f, 0.f);
}

// ------------------------------------------------- edge scatter-add (segment_sum)
// thread -> (edge, 4-feature chunk); float4 gather (L2-resident), 4x atomic f32 add
template<int D>
__global__ void scatter_add_kernel(const float* __restrict__ x,
                                   const int* __restrict__ src,
                                   const int* __restrict__ dst,
                                   float* __restrict__ agg,
                                   int E) {
  constexpr int PER = D >> 2;                    // 16 or 32 chunks per edge
  long gid = (long)blockIdx.x * blockDim.x + threadIdx.x;
  long e = gid / PER;                            // PER is a power of two -> shift
  if (e >= E) return;
  int c = (int)(gid & (PER - 1)) << 2;
  int s = src[e];
  int d = dst[e];
  float4 v = *(const float4*)(x + (size_t)s * D + c);
  float* o = agg + (size_t)d * D + c;
  atomicAdd(o + 0, v.x);
  atomicAdd(o + 1, v.y);
  atomicAdd(o + 2, v.z);
  atomicAdd(o + 3, v.w);
}

// ------------------------------------------ fused WMMA concat-GEMM + L2norm + BN
// out[N,DOUT] = postproc( [A0|A1] (N x 2*KHALF) @ [W0;W1] + bias )
//   postproc = row-L2-normalize, then eval-mode BN (folded a*h+b), optional ReLU.
// 256 threads = 8 waves; wave owns a 16-row strip. D-tile goes through padded LDS
// so the final global stores are coalesced float4/float2 rows (no strided b32s).
template<int KHALF, int DOUT, bool RELU>
__global__ void __launch_bounds__(256)
gemm_concat_norm_bn(const float* __restrict__ A0, const float* __restrict__ A1,
                    const float* __restrict__ W0, const float* __restrict__ W1,
                    const float* __restrict__ bias,
                    const float* __restrict__ gamma, const float* __restrict__ beta,
                    const float* __restrict__ rm, const float* __restrict__ rv,
                    float* __restrict__ out, int Nrows) {
  constexpr int KTOT   = KHALF * 2;
  constexpr int KPAD   = KTOT + 8;               // +16B: rotate LDS banks per column
  constexpr int NKS    = KTOT / 32;              // 4 (layer1) or 8 (layer2)
  constexpr int NTILES = DOUT / 16;              // 8 (layer1) or 4 (layer2)
  constexpr int DPAD   = DOUT + 4;               // dword pad: kill row-bank conflicts
  constexpr int CPL    = DOUT / 32;              // cols per lane in epilogue (4 or 2)

  __shared__ __attribute__((aligned(16))) unsigned short Wt[DOUT * KPAD];
  __shared__ __attribute__((aligned(16))) float Dtile[8 * 16 * DPAD];

  const int tid = threadIdx.x;

  // stage weights transposed + converted to bf16 (tiny: 64KB of f32, L2-hot)
  for (int idx = tid; idx < KTOT * DOUT; idx += 256) {
    int k = idx / DOUT;
    int c = idx - k * DOUT;
    float w = (k < KHALF) ? W0[(size_t)k * DOUT + c]
                          : W1[(size_t)(k - KHALF) * DOUT + c];
    Wt[c * KPAD + k] = f2bf(w);
  }
  __syncthreads();

  const int wave = tid >> 5;
  const int lane = tid & 31;
  const int row_base = blockIdx.x * 128 + wave * 16;
  const int m    = lane & 15;
  const int half = lane >> 4;                    // 0 or 1
  int arow = row_base + m;
  if (arow >= Nrows) arow = Nrows - 1;           // clamp loads; stores row-guarded

  // A fragments for all K steps, converted f32 -> bf16 on the fly.
  // Layout (16-bit A 16x32): lanes 0-15 K{0-7,16-23}, lanes 16-31 K{8-15,24-31}.
  const int klo = half * 8;
  Frag16 afrag[NKS];
  #pragma unroll
  for (int ks = 0; ks < NKS; ++ks) {
    int kb = ks * 32;
    const float* srcp = (kb < KHALF) ? (A0 + (size_t)arow * KHALF + kb)
                                     : (A1 + (size_t)arow * KHALF + (kb - KHALF));
    float4 q0 = *(const float4*)(srcp + klo);
    float4 q1 = *(const float4*)(srcp + klo + 4);
    float4 q2 = *(const float4*)(srcp + 16 + klo);
    float4 q3 = *(const float4*)(srcp + 16 + klo + 4);
    Frag16 f;
    f.s[0]  = f2bf(q0.x); f.s[1]  = f2bf(q0.y); f.s[2]  = f2bf(q0.z); f.s[3]  = f2bf(q0.w);
    f.s[4]  = f2bf(q1.x); f.s[5]  = f2bf(q1.y); f.s[6]  = f2bf(q1.z); f.s[7]  = f2bf(q1.w);
    f.s[8]  = f2bf(q2.x); f.s[9]  = f2bf(q2.y); f.s[10] = f2bf(q2.z); f.s[11] = f2bf(q2.w);
    f.s[12] = f2bf(q3.x); f.s[13] = f2bf(q3.y); f.s[14] = f2bf(q3.z); f.s[15] = f2bf(q3.w);
    afrag[ks] = f;
  }

  // B fragment (32x16 bf16): lane n -> column n, K {0-7,16-23} (lanes 0-15) / {8-15,24-31} (16-31)
  float* dt = &Dtile[wave * 16 * DPAD];          // per-wave 16 x DOUT f32 tile
  const int koff = half * 8;
  #pragma unroll
  for (int ct = 0; ct < NTILES; ++ct) {
    int col = (ct << 4) + m;
    float bv = bias[col];
    v8f acc = {bv, bv, bv, bv, bv, bv, bv, bv};
    const unsigned short* wrow = &Wt[col * KPAD + koff];
    #pragma unroll
    for (int ks = 0; ks < NKS; ++ks) {
      Frag16 b;
      const uint4* bp = (const uint4*)(wrow + (ks << 5));   // 16B aligned
      b.q[0] = bp[0];
      b.q[1] = bp[2];
      acc = __builtin_amdgcn_wmma_f32_16x16x32_bf16(
          /*neg_a=*/false, afrag[ks].v, /*neg_b=*/false, b.v,
          /*c_mod=*/(short)0, acc, /*reuse_a=*/false, /*reuse_b=*/false);
    }
    // D layout: VGPR r -> M = r + 8*half, N = col. Dump to per-wave LDS tile.
    int rlo = half * 8;
    #pragma unroll
    for (int r = 0; r < 8; ++r) dt[(rlo + r) * DPAD + col] = acc[r];
  }

  // ---- fused epilogue: per row -> L2 normalize, folded BN, optional ReLU ----
  // Fold BN: o = h * (gamma*rsqrt(rv+eps)) + (beta - gamma*rsqrt(rv+eps)*rm)
  const int c0 = lane * CPL;
  float ga[CPL], bb[CPL];
  #pragma unroll
  for (int j = 0; j < CPL; ++j) {
    int c = c0 + j;
    float rs = rsqrtf(rv[c] + EPS_BN);
    ga[j] = gamma[c] * rs;
    bb[j] = beta[c] - ga[j] * rm[c];
  }

  for (int r = 0; r < 16; ++r) {
    int row = row_base + r;
    if (row >= Nrows) break;                     // wave-uniform
    const float* drow = dt + r * DPAD;
    float vv[CPL];
    float ss = 0.f;
    if constexpr (CPL == 4) {
      float4 q = *(const float4*)(drow + c0);
      vv[0] = q.x; vv[1] = q.y; vv[2] = q.z; vv[3] = q.w;
    } else {
      float2 q = *(const float2*)(drow + c0);
      vv[0] = q.x; vv[1] = q.y;
    }
    #pragma unroll
    for (int j = 0; j < CPL; ++j) ss += vv[j] * vv[j];
    #pragma unroll
    for (int off = 16; off >= 1; off >>= 1) ss += __shfl_xor(ss, off, 32);
    float sc = 1.0f / fmaxf(sqrtf(ss), EPS_NORM);
    float o[CPL];
    #pragma unroll
    for (int j = 0; j < CPL; ++j) {
      float v = ga[j] * (vv[j] * sc) + bb[j];
      o[j] = RELU ? fmaxf(v, 0.f) : v;
    }
    float* orow = out + (size_t)row * DOUT;
    if constexpr (CPL == 4) {
      *(float4*)(orow + c0) = make_float4(o[0], o[1], o[2], o[3]);
    } else {
      *(float2*)(orow + c0) = make_float2(o[0], o[1]);
    }
  }
}

// ---------------------------------------------------------------- launcher
extern "C" void kernel_launch(void* const* d_in, const int* in_sizes, int n_in,
                              void* d_out, int out_size, void* d_ws, size_t ws_size,
                              hipStream_t stream) {
  const float* x   = (const float*)d_in[0];
  const int*   ei  = (const int*)d_in[1];
  const float* W1l = (const float*)d_in[2];
  const float* b1  = (const float*)d_in[3];
  const float* W1r = (const float*)d_in[4];
  const float* g1  = (const float*)d_in[5];
  const float* be1 = (const float*)d_in[6];
  const float* rm1 = (const float*)d_in[7];
  const float* rv1 = (const float*)d_in[8];
  const float* W2l = (const float*)d_in[9];
  const float* b2  = (const float*)d_in[10];
  const float* W2r = (const float*)d_in[11];
  const float* g2  = (const float*)d_in[12];
  const float* be2 = (const float*)d_in[13];
  const float* rm2 = (const float*)d_in[14];
  const float* rv2 = (const float*)d_in[15];

  const int D_IN = 64, D_H = 128;
  const int N = in_sizes[0] / D_IN;
  const int E = in_sizes[1] / 2;
  const int* src = ei;
  const int* dst = ei + E;

  // workspace: agg1 (N*64) | h (N*128) | agg2 (N*128)  = 64 MB
  float* agg1 = (float*)d_ws;
  float* h    = agg1 + (size_t)N * D_IN;
  float* agg2 = h + (size_t)N * D_H;
  float* out  = (float*)d_out;

  // ---- layer 1 ----
  { long n4 = (long)N * D_IN / 4;
    zero_kernel<<<(int)((n4 + 255) / 256), 256, 0, stream>>>((float4*)agg1, n4); }
  { long t = (long)E * (D_IN / 4);
    scatter_add_kernel<64><<<(int)((t + 255) / 256), 256, 0, stream>>>(x, src, dst, agg1, E); }
  { int blocks = (N + 127) / 128;
    gemm_concat_norm_bn<64, 128, true><<<blocks, 256, 0, stream>>>(
        agg1, x, W1l, W1r, b1, g1, be1, rm1, rv1, h, N); }

  // ---- layer 2 ----
  { long n4 = (long)N * D_H / 4;
    zero_kernel<<<(int)((n4 + 255) / 256), 256, 0, stream>>>((float4*)agg2, n4); }
  { long t = (long)E * (D_H / 4);
    scatter_add_kernel<128><<<(int)((t + 255) / 256), 256, 0, stream>>>(h, src, dst, agg2, E); }
  { int blocks = (N + 127) / 128;
    gemm_concat_norm_bn<128, 64, false><<<blocks, 256, 0, stream>>>(
        agg2, h, W2l, W2r, b2, g2, be2, rm2, rv2, out, N); }
}
